// GCNWildfire_70772471103954
// MI455X (gfx1250) — matrix operand order repaired and checked by reading, not compiled, not run on hardware
//
#include <hip/hip_runtime.h>
#include <hip/hip_bf16.h>

// ---------------------------------------------------------------------------
// GCN forward on MI455X (gfx1250, wave32).
// GEMMs via v_wmma_f32_16x16x32_bf16 (f32 accumulate). B (weights) are
// pre-swizzled once into WMMA-fragment order in global memory -> fragment
// loads are two global_load_b128 per lane, L2-resident. A tiles staged via
// double-buffered LDS (one barrier per K-step) with branchless row-clamped
// loads. Scatter via global_atomic_add_f32 (102MB node buffers < 192MB L2).
// Linear bias absorbed by BatchNorm (exact).
// ---------------------------------------------------------------------------

typedef __attribute__((ext_vector_type(16))) __bf16 v16bf;
typedef __attribute__((ext_vector_type(8)))  __bf16 v8bf;
typedef __attribute__((ext_vector_type(4)))  __bf16 v4bf;
typedef __attribute__((ext_vector_type(8)))  float  v8f;

#define N_NODES 100000
#define N_EDGES 1600000
#define IN_CH   61
#define HID     256

// -------- weight cast: fp32 -> bf16, swizzled into WMMA B-fragment order ----
// Layout per K-tile kt (32 rows), col-group ct (16 cols), lane (0..31):
//   frag[kt][ct][lane][i] = W[kt*32 + (lane>>4)*16 + i][ct*16 + (lane&15)]
// so each lane's 16-half fragment is contiguous (32B).
__global__ void k_castw(const float* __restrict__ inW,
                        const float* __restrict__ convW,
                        __bf16* __restrict__ inWb,
                        __bf16* __restrict__ convWb) {
    int gid = blockIdx.x * blockDim.x + threadIdx.x;
    if (gid < 2 * 16 * 512) {                       // input: K padded to 64
        int i = gid & 15, lane = (gid >> 4) & 31;
        int ct = (gid >> 9) & 15, kt = gid >> 13;
        int k = kt * 32 + (lane >> 4) * 16 + i;
        int n = ct * 16 + (lane & 15);
        inWb[gid] = (k < IN_CH) ? (__bf16)inW[(size_t)k * 256 + n] : (__bf16)0.0f;
    }
    int g2 = gid - 2 * 16 * 512;
    if (g2 >= 0 && g2 < 4 * 8 * 16 * 512) {         // 4 conv layers, K=256
        int idx = g2 & 65535, l = g2 >> 16;
        int i = idx & 15, lane = (idx >> 4) & 31;
        int ct = (idx >> 9) & 15, kt = idx >> 13;
        int k = kt * 32 + (lane >> 4) * 16 + i;
        int n = ct * 16 + (lane & 15);
        convWb[g2] = (__bf16)convW[(size_t)l * 65536 + (size_t)k * 256 + n];
    }
}

// ---------------- fills ------------------------------------------------------
__global__ void k_fill(float* __restrict__ p, long long n, float v) {
    long long i = (long long)blockIdx.x * blockDim.x + threadIdx.x;
    if (i < n) p[i] = v;
}
__global__ void k_zero4(float4* __restrict__ p, long long n4) {
    long long i = (long long)blockIdx.x * blockDim.x + threadIdx.x;
    if (i < n4) p[i] = make_float4(0.f, 0.f, 0.f, 0.f);
}

// ---------------- degree / dinv / edge norm ---------------------------------
__global__ void k_deg(const long long* __restrict__ ei, float* __restrict__ deg,
                      int nEdges) {
    int e = blockIdx.x * blockDim.x + threadIdx.x;
    if (e < nEdges) unsafeAtomicAdd(&deg[(int)ei[nEdges + e]], 1.0f);
}
__global__ void k_dinv(const float* __restrict__ deg, float* __restrict__ dinv,
                       int n) {
    int i = blockIdx.x * blockDim.x + threadIdx.x;
    if (i < n) dinv[i] = rsqrtf(deg[i]);  // deg >= 1 (self loop)
}
__global__ void k_norm(const long long* __restrict__ ei,
                       const float* __restrict__ dinv, float* __restrict__ nv,
                       int nEdges, int nNodes) {
    int e = blockIdx.x * blockDim.x + threadIdx.x;
    if (e < nEdges) {
        nv[e] = dinv[(int)ei[e]] * dinv[(int)ei[nEdges + e]];
    } else if (e < nEdges + nNodes) {
        float d = dinv[e - nEdges];
        nv[e] = d * d;
    }
}

// ---------------- WMMA GEMM: C[N x 256] = A[N x KEFF] @ B[K x 256] ----------
// Block: 256 threads = 8 waves. Tile: M=32, N=256. Wave w: mg=w>>2 (M sub),
// ng=w&3 (64-col group -> 4x 16x16 accumulators). B comes pre-swizzled.
// Template params compile the conv path branch-free; A tile double-buffered.
template <int KTILES, int LDA, int KEFF>
__global__ __launch_bounds__(256)
void k_gemm_bf16(const float* __restrict__ A, const __bf16* __restrict__ B,
                 float* __restrict__ C, int nRows) {
    __shared__ __bf16 As[2][32][40];   // 2x (32x32 bf16), stride 40: 16B frags

    const int tid  = threadIdx.x;
    const int lane = tid & 31;
    const int wave = tid >> 5;
    const int mg   = wave >> 2;        // 0..1
    const int ng   = wave & 3;         // 0..3
    const int half = lane >> 4;        // 0..1
    const int lq   = lane & 15;        // 0..15
    const int rowBase = blockIdx.x * 32;

    const int ldr = tid >> 3;          // 0..31 (tile row this thread stages)
    const int ak0 = (tid & 7) * 4;     // 4 A columns per thread
    // branchless row clamp: out-of-range rows feed only unstored C rows
    const int crow = min(rowBase + ldr, nRows - 1);
    const float* __restrict__ arow = A + (size_t)crow * LDA;

    v8f acc[4] = {};

    auto stage = [&](int t) {
        const int kk = t * 32;
        v4bf h4;
        if ((LDA & 3) == 0 && kk + 32 <= KEFF) {       // compile-time for conv
            float4 v = *(const float4*)(arow + kk + ak0);
            h4[0] = (__bf16)v.x; h4[1] = (__bf16)v.y;
            h4[2] = (__bf16)v.z; h4[3] = (__bf16)v.w;
        } else {                                       // K-tail (input GEMM)
            #pragma unroll
            for (int j = 0; j < 4; ++j) {
                int k = kk + ak0 + j;
                float v = arow[min(k, KEFF - 1)];
                h4[j] = (__bf16)((k < KEFF) ? v : 0.0f);
            }
        }
        *(v4bf*)&As[t & 1][ldr][ak0] = h4;             // one ds_store_b64
        if ((t + 1) * 32 < KEFF) __builtin_prefetch(arow + (t + 1) * 32, 0, 3);
    };

    stage(0);
    __syncthreads();

    #pragma unroll
    for (int t = 0; t < KTILES; ++t) {
        if (t + 1 < KTILES) stage(t + 1);              // fill other buffer

        // --- A fragment: two 16B LDS chunks -> v16bf (k = half*16 + s, same order as B) ---
        const __bf16* ap = &As[t & 1][mg * 16 + lq][half * 16];
        v8bf alo = *(const v8bf*)ap;
        v8bf ahi = *(const v8bf*)(ap + 8);
        v16bf af = __builtin_shufflevector(alo, ahi, 0, 1, 2, 3, 4, 5, 6, 7,
                                           8, 9, 10, 11, 12, 13, 14, 15);

        // --- all 4 B fragments up front (one clause of global_load_b128) ---
        const __bf16* bt = B + ((size_t)t * 16 + ng * 4) * 512
                             + (size_t)lane * 16;
        v16bf bfrag[4];
        #pragma unroll
        for (int nt = 0; nt < 4; ++nt) {
            v8bf blo = *(const v8bf*)(bt + (size_t)nt * 512);
            v8bf bhi = *(const v8bf*)(bt + (size_t)nt * 512 + 8);
            bfrag[nt] = __builtin_shufflevector(blo, bhi, 0, 1, 2, 3, 4, 5, 6,
                                                7, 8, 9, 10, 11, 12, 13, 14, 15);
        }
        // --- 4 WMMAs ---
        #pragma unroll
        for (int nt = 0; nt < 4; ++nt)
            acc[nt] = __builtin_amdgcn_wmma_f32_16x16x32_bf16(
                false, af, false, bfrag[nt], (short)0, acc[nt], false, false);

        __syncthreads();                               // single barrier/K-step
    }

    // --- epilogue: C/D layout M = half*8 + r, N = lq ---
    const int mrow = rowBase + mg * 16 + half * 8;
    #pragma unroll
    for (int r = 0; r < 8; ++r) {
        int grow = mrow + r;
        if (grow < nRows) {
            #pragma unroll
            for (int nt = 0; nt < 4; ++nt)
                C[(size_t)grow * 256 + ng * 64 + nt * 16 + lq] = acc[nt][r];
        }
    }
}

// ---------------- edge scatter: agg[dst] += xw[src] * norm ------------------
// One thread = 4 channels of one edge (64 threads/edge).
__global__ void k_scatter(const long long* __restrict__ ei,
                          const float* __restrict__ nv,
                          const float* __restrict__ xw,
                          float* __restrict__ agg, int nEdges, int nNodes) {
    long long gid = (long long)blockIdx.x * blockDim.x + threadIdx.x;
    long long e = gid >> 6;
    if (e >= (long long)nEdges + nNodes) return;
    int c = (int)(gid & 63) * 4;
    int s, d;
    if (e < nEdges) {
        s = (int)ei[e];
        d = (int)ei[nEdges + e];
    } else {
        s = d = (int)(e - nEdges);
    }
    float w = nv[e];
    float4 v = *(const float4*)(xw + (size_t)s * 256 + c);
    float* ap = agg + (size_t)d * 256 + c;
    unsafeAtomicAdd(ap + 0, v.x * w);
    unsafeAtomicAdd(ap + 1, v.y * w);
    unsafeAtomicAdd(ap + 2, v.z * w);
    unsafeAtomicAdd(ap + 3, v.w * w);
}

// ---------------- BN batch stats (sum, sumsq per channel) -------------------
__global__ void k_stats(const float* __restrict__ X, int nRows,
                        float* __restrict__ stats) {
    int c = threadIdx.x;                 // channel 0..255
    int r0 = blockIdx.x * 256;
    int rend = min(r0 + 256, nRows);
    float s = 0.f, s2 = 0.f;
    for (int r = r0; r < rend; ++r) {
        float v = X[(size_t)r * 256 + c];
        s += v; s2 += v * v;
    }
    unsafeAtomicAdd(&stats[c], s);
    unsafeAtomicAdd(&stats[256 + c], s2);
}

__global__ void k_bnfin(const float* __restrict__ stats,
                        const float* __restrict__ gamma,
                        const float* __restrict__ beta,
                        float* __restrict__ ss, int nRows) {
    int c = threadIdx.x;
    float inv = 1.0f / (float)nRows;
    float mu  = stats[c] * inv;
    float var = stats[256 + c] * inv - mu * mu;
    float sc  = gamma[c] * rsqrtf(var + 1e-5f);
    ss[c] = sc;
    ss[256 + c] = beta[c] - mu * sc;
}

// ---------------- apply: Y = relu(sc*X + sh) (+ residual) -------------------
__global__ void k_apply(const float* __restrict__ X,
                        const float* __restrict__ ss,
                        const float* __restrict__ res,
                        float* __restrict__ Y, int nRows) {
    long long gid = (long long)blockIdx.x * blockDim.x + threadIdx.x;
    long long total = (long long)nRows * 64;     // float4 granules
    if (gid >= total) return;
    int c = (int)(gid & 63) * 4;
    float4 v = *(const float4*)(X + gid * 4);
    float4 o;
    o.x = fmaxf(0.f, v.x * ss[c + 0] + ss[256 + c + 0]);
    o.y = fmaxf(0.f, v.y * ss[c + 1] + ss[256 + c + 1]);
    o.z = fmaxf(0.f, v.z * ss[c + 2] + ss[256 + c + 2]);
    o.w = fmaxf(0.f, v.w * ss[c + 3] + ss[256 + c + 3]);
    if (res) {
        float4 r = *(const float4*)(res + gid * 4);
        o.x += r.x; o.y += r.y; o.z += r.z; o.w += r.w;
    }
    *(float4*)(Y + gid * 4) = o;
}

// ---------------- output heads: mean / clipped log_var ----------------------
__global__ void k_heads(const float* __restrict__ Hb,
                        const float* __restrict__ mW, const float* __restrict__ mb,
                        const float* __restrict__ lW, const float* __restrict__ lb,
                        float* __restrict__ out, int nNodes) {
    int wave = threadIdx.x >> 5, lane = threadIdx.x & 31;
    int node = blockIdx.x * 8 + wave;
    if (node >= nNodes) return;                 // uniform per wave
    const float* h = Hb + (size_t)node * 256;
    float sm = 0.f, sl = 0.f;
    #pragma unroll
    for (int i = 0; i < 256; i += 32) {
        float v = h[i + lane];
        sm += v * mW[i + lane];
        sl += v * lW[i + lane];
    }
    #pragma unroll
    for (int off = 16; off; off >>= 1) {
        sm += __shfl_xor(sm, off, 32);
        sl += __shfl_xor(sl, off, 32);
    }
    if (lane == 0) {
        out[node] = sm + mb[0];
        float lv = sl + lb[0];
        out[nNodes + node] = fminf(10.f, fmaxf(-10.f, lv));
    }
}

// ---------------------------------------------------------------------------
extern "C" void kernel_launch(void* const* d_in, const int* in_sizes, int n_in,
                              void* d_out, int out_size, void* d_ws, size_t ws_size,
                              hipStream_t stream) {
    const float*     x        = (const float*)d_in[0];
    const long long* ei       = (const long long*)d_in[1];   // int64 edge_index
    const float*     in_W     = (const float*)d_in[2];
    /* in_b absorbed by BN */
    const float*     in_gamma = (const float*)d_in[4];
    const float*     in_beta  = (const float*)d_in[5];
    const float*     conv_W   = (const float*)d_in[6];
    /* conv_b absorbed by BN */
    const float*     bn_gamma = (const float*)d_in[8];
    const float*     bn_beta  = (const float*)d_in[9];
    const float*     mean_W   = (const float*)d_in[10];
    const float*     mean_b   = (const float*)d_in[11];
    const float*     lv_W     = (const float*)d_in[12];
    const float*     lv_b     = (const float*)d_in[13];
    float* out = (float*)d_out;

    const int N = N_NODES, E = N_EDGES;
    const long long NB = (long long)N * HID;     // elements per node buffer

    // workspace carve-up
    float* buf[3];
    buf[0] = (float*)d_ws;
    buf[1] = buf[0] + NB;
    buf[2] = buf[1] + NB;
    float*  deg    = buf[2] + NB;
    float*  dinv   = deg + N;
    float*  normv  = dinv + N;                   // E + N
    float*  stats  = normv + (E + N);            // 512
    float*  ss     = stats + 512;                // 512
    __bf16* inWb   = (__bf16*)(ss + 512);        // 64*256 (swizzled)
    __bf16* convWb = inWb + 64 * 256;            // 4*256*256 (swizzled)

    // --- precompute: swizzled bf16 weights + symmetric GCN edge norms ---
    k_castw<<<(64 * 256 + 4 * 256 * 256 + 255) / 256, 256, 0, stream>>>(
        in_W, conv_W, inWb, convWb);
    k_fill<<<(N + 255) / 256, 256, 0, stream>>>(deg, N, 1.0f);
    k_deg<<<(E + 255) / 256, 256, 0, stream>>>(ei, deg, E);
    k_dinv<<<(N + 255) / 256, 256, 0, stream>>>(deg, dinv, N);
    k_norm<<<(E + N + 255) / 256, 256, 0, stream>>>(ei, dinv, normv, E, N);

    const int gemmBlocks  = (N + 31) / 32;
    const int applyBlocks = (int)((NB / 4 + 255) / 256);
    const int statsBlocks = (N + 255) / 256;
    const long long scatWork = ((long long)(E + N) * 64 + 255) / 256;

    // --- input projection: x @ in_W -> BN -> ReLU  (H in buf[0]) ---
    k_gemm_bf16<2, IN_CH, IN_CH><<<gemmBlocks, 256, 0, stream>>>(x, inWb,
                                                                 buf[1], N);
    k_fill<<<2, 256, 0, stream>>>(stats, 512, 0.0f);
    k_stats<<<statsBlocks, 256, 0, stream>>>(buf[1], N, stats);
    k_bnfin<<<1, 256, 0, stream>>>(stats, in_gamma, in_beta, ss, N);
    k_apply<<<applyBlocks, 256, 0, stream>>>(buf[1], ss, nullptr, buf[0], N);

    // --- 4 GCN layers, rotating (H, XW, AGG) through the 3 buffers ---
    int hb = 0, xb = 1, ab = 2;
    for (int l = 0; l < 4; ++l) {
        k_gemm_bf16<8, HID, HID><<<gemmBlocks, 256, 0, stream>>>(
            buf[hb], convWb + (size_t)l * HID * HID, buf[xb], N);
        k_zero4<<<(int)((NB / 4 + 255) / 256), 256, 0, stream>>>(
            (float4*)buf[ab], NB / 4);
        k_scatter<<<(int)scatWork, 256, 0, stream>>>(ei, normv, buf[xb],
                                                     buf[ab], E, N);
        k_fill<<<2, 256, 0, stream>>>(stats, 512, 0.0f);
        k_stats<<<statsBlocks, 256, 0, stream>>>(buf[ab], N, stats);
        k_bnfin<<<1, 256, 0, stream>>>(stats, bn_gamma + (size_t)l * HID,
                                       bn_beta + (size_t)l * HID, ss, N);
        k_apply<<<applyBlocks, 256, 0, stream>>>(buf[ab], ss, buf[hb],
                                                 buf[xb], N);
        int t = hb; hb = xb; xb = ab; ab = t;    // rotate roles
    }

    // --- heads ---
    k_heads<<<(N + 7) / 8, 256, 0, stream>>>(buf[hb], mean_W, mean_b, lv_W,
                                             lv_b, out, N);
}
